// MoETransformerBlock_25434796327322
// MI455X (gfx1250) — matrix operand order repaired and checked by reading, not compiled, not run on hardware
//
#include <hip/hip_runtime.h>
#include <cstdint>
#include <cstddef>

// ---------------------------------------------------------------------------
// MoE transformer block for MI455X (gfx1250, wave32, WMMA).
// Heavy math: V_WMMA_F32_16X16X32_BF16 (bf16 in, f32 acc).
// Data movement: ping-pong LDS pipeline -- GLOBAL_LOAD_ASYNC_TO_LDS_B128
// (ASYNCcnt) issued for tile i+1 while tile i runs on the WMMA pipe;
// ds_store_b128 for transpose/convert staging; b128 LDS fragment loads.
// ---------------------------------------------------------------------------

typedef __bf16 v16bf __attribute__((ext_vector_type(16)));
typedef float  v8f   __attribute__((ext_vector_type(8)));

static __device__ __forceinline__ unsigned short f2bf(float f) {
  unsigned int u = __float_as_uint(f);
  unsigned int r = u + 0x7fffu + ((u >> 16) & 1u);   // round-to-nearest-even
  return (unsigned short)(r >> 16);
}
static __device__ __forceinline__ float bf2f(unsigned short h) {
  return __uint_as_float(((unsigned int)h) << 16);
}

union Frag {
  v16bf v;
  unsigned short u16[16];
  unsigned int   u32[8];
};

// Issue one per-lane 16-byte async copy global->LDS (tracked by ASYNCcnt).
static __device__ __forceinline__ void async_b128(unsigned int lds_off,
                                                  const void* gsrc) {
  unsigned long long g = (unsigned long long)(uintptr_t)gsrc;
  asm volatile("global_load_async_to_lds_b128 %0, %1, off"
               :: "v"(lds_off), "v"(g)
               : "memory");
}
static __device__ __forceinline__ void wait_async0() {
  asm volatile("s_wait_asynccnt 0x0" ::: "memory");
}
// Low 32 bits of a generic pointer into LDS == LDS byte offset on AMDGPU.
static __device__ __forceinline__ unsigned int lds_off32(const void* p) {
  return (unsigned int)(uintptr_t)p;
}

// ---------------------------------------------------------------------------
// Generic tiled WMMA GEMM:  C[M,N] = epilogue( A(bf16)[M,K] * B[K,N] )
// BMODE 0: B is bf16 row-major [K,N]  (staged with transpose to K-major LDS)
// BMODE 1: B is f32  row-major [K,N]  (converted to bf16 while staging)
// BMODE 2: B is bf16 row-major [N,K]  (contiguous: async copy straight to LDS)
// mode 0: Cf = alpha*acc
// mode 1: Cbf = bf16(alpha*acc)
// mode 2: Cf = acc + residual
// mode 3: Cf += rowScale[row*ss] * acc
// mode 4: Cf = residual + rowScale[row*ss] * acc
// Assumes M % 128 == 0 and K % 32 == 0 (true for all call sites); N guarded.
// ---------------------------------------------------------------------------
#define BM 128
#define BN 128
#define BKT 32
#define BKP (BKT + 8)   // padded K stride (bf16 elems): 80B rows, 16B aligned

template <int BMODE>
__global__ __launch_bounds__(256) void gemm_wmma(
    const unsigned short* __restrict__ A, int lda,
    const void* __restrict__ Bp, int ldb,
    int M, int N, int Kd, float alpha,
    float* __restrict__ Cf, unsigned short* __restrict__ Cbf, int ldc,
    const float* __restrict__ residual,
    const float* __restrict__ rowScale, int scaleStride,
    int mode) {
  // Ping-pong, K-major tiles with padded stride: [buf][row|col][BKP]
  __shared__ unsigned short lds_a[2][BM * BKP];
  __shared__ unsigned short lds_b[2][BN * BKP];

  const int tid  = threadIdx.x;
  const int lane = tid & 31;
  const int wave = tid >> 5;
  const int wm   = (wave & 1) * 64;   // wave's M offset within the block tile
  const int wn   = (wave >> 1) * 32;  // wave's N offset within the block tile
  const long blockM = (long)blockIdx.y * BM;
  const long blockN = (long)blockIdx.x * BN;
  const int hf  = lane >> 4;          // wmma lane half
  const int l16 = lane & 15;

  v8f acc[4][2];
#pragma unroll
  for (int mi = 0; mi < 4; ++mi)
#pragma unroll
    for (int ni = 0; ni < 2; ++ni)
#pragma unroll
      for (int r = 0; r < 8; ++r) acc[mi][ni][r] = 0.0f;

  const int arow = tid >> 1;          // A-stage: row 0..127
  const int akb  = (tid & 1) << 4;    // A-stage: K sub-offset 0/16
  const int bn   = tid >> 1;          // B-stage: column 0..127
  const int bkb  = (tid & 1) << 4;    // B-stage: K sub-offset 0/16
  const long gn  = blockN + bn;
  const bool gnOK = gn < (long)N;

  // Issue async copies for one K-tile (A always; B only in BMODE 2).
  auto issue_async = [&](int k0, int buf) {
    const unsigned short* asrc = A + (blockM + arow) * (long)lda + k0 + akb;
    unsigned int la = lds_off32(&lds_a[buf][arow * BKP + akb]);
    async_b128(la, asrc);
    async_b128(la + 16, asrc + 8);
    if (BMODE == 2) {
      unsigned int lb = lds_off32(&lds_b[buf][bn * BKP + bkb]);
      if (gnOK) {
        const unsigned short* bsrc =
            (const unsigned short*)Bp + gn * (long)ldb + k0 + bkb;
        async_b128(lb, bsrc);
        async_b128(lb + 16, bsrc + 8);
      } else {
        uint4 z = {0u, 0u, 0u, 0u};
        *(uint4*)(&lds_b[buf][bn * BKP + bkb]) = z;
        *(uint4*)(&lds_b[buf][bn * BKP + bkb + 8]) = z;
      }
    }
  };
  // Gather one K-tile of B into registers (BMODE 0/1): coalesced across lanes.
  auto gather_b = [&](int k0, unsigned int u[8]) {
    if (BMODE == 1) {
      const float* bsrc = (const float*)Bp + (long)(k0 + bkb) * ldb + gn;
      if (gnOK) {
#pragma unroll
        for (int j = 0; j < 8; ++j) {
          float f0 = bsrc[(2 * j) * (long)ldb];
          float f1 = bsrc[(2 * j + 1) * (long)ldb];
          u[j] = ((unsigned int)f2bf(f1) << 16) | (unsigned int)f2bf(f0);
        }
        if (k0 + BKT < Kd) {  // hint next K-tile (global_prefetch_b8)
          __builtin_prefetch(bsrc + (long)BKT * ldb, 0, 0);
          __builtin_prefetch(bsrc + (long)(BKT + 8) * ldb, 0, 0);
        }
      } else {
#pragma unroll
        for (int j = 0; j < 8; ++j) u[j] = 0u;
      }
    } else if (BMODE == 0) {
      const unsigned short* bsrc =
          (const unsigned short*)Bp + (long)(k0 + bkb) * ldb + gn;
      if (gnOK) {
#pragma unroll
        for (int j = 0; j < 8; ++j) {
          unsigned int lo = bsrc[(2 * j) * (long)ldb];
          unsigned int hi = bsrc[(2 * j + 1) * (long)ldb];
          u[j] = (hi << 16) | lo;
        }
      } else {
#pragma unroll
        for (int j = 0; j < 8; ++j) u[j] = 0u;
      }
    }
  };
  auto store_b = [&](int buf, const unsigned int u[8]) {
    uint4* dst = (uint4*)(&lds_b[buf][bn * BKP + bkb]);
    dst[0] = make_uint4(u[0], u[1], u[2], u[3]);
    dst[1] = make_uint4(u[4], u[5], u[6], u[7]);
  };
  // 12 x ds_load_b128 -> 8 x v_wmma per call.
  auto compute = [&](int buf) {
    Frag afr[4], bfr[2];
#pragma unroll
    for (int mi = 0; mi < 4; ++mi) {
      int r = wm + mi * 16 + l16;
      const unsigned int* base = (const unsigned int*)(&lds_a[buf][r * BKP]);
#pragma unroll
      for (int j = 0; j < 8; ++j) {
        int kk = 2 * j + 16 * hf;  // K=16*hf..+15
        afr[mi].u32[j] = base[kk >> 1];
      }
    }
#pragma unroll
    for (int ni = 0; ni < 2; ++ni) {
      int n = wn + ni * 16 + l16;
      const unsigned int* base =
          (const unsigned int*)(&lds_b[buf][n * BKP + 16 * hf]);
#pragma unroll
      for (int j = 0; j < 8; ++j) bfr[ni].u32[j] = base[j];  // K=16*hf..+15
    }
#pragma unroll
    for (int mi = 0; mi < 4; ++mi)
#pragma unroll
      for (int ni = 0; ni < 2; ++ni)
        acc[mi][ni] = __builtin_amdgcn_wmma_f32_16x16x32_bf16(
            false, afr[mi].v, false, bfr[ni].v, (short)0, acc[mi][ni], false,
            false);
  };

  const int ntiles = Kd / BKT;

  // ---- prologue: fill buffer 0 ----
  {
    issue_async(0, 0);
    unsigned int u[8];
    if (BMODE != 2) {
      gather_b(0, u);
      store_b(0, u);
    }
    wait_async0();
    __syncthreads();
  }

  // ---- pipelined main loop: stage t+1 while computing t ----
  for (int t = 0; t < ntiles; ++t) {
    const int buf  = t & 1;
    const int nbuf = (t + 1) & 1;
    const bool more = (t + 1) < ntiles;
    unsigned int u[8];
    if (more) {
      issue_async((t + 1) * BKT, nbuf);       // async engine runs under WMMA
      if (BMODE != 2) gather_b((t + 1) * BKT, u);
    }
    compute(buf);
    if (more) {
      if (BMODE != 2) store_b(nbuf, u);
      wait_async0();                          // tile t+1 landed in LDS
      __syncthreads();
    }
  }

  // ---- epilogue ----
#pragma unroll
  for (int mi = 0; mi < 4; ++mi) {
#pragma unroll
    for (int ni = 0; ni < 2; ++ni) {
#pragma unroll
      for (int r = 0; r < 8; ++r) {
        long row = blockM + wm + mi * 16 + r + 8 * hf;
        long col = blockN + wn + ni * 16 + l16;
        if (col >= (long)N) continue;
        float a = acc[mi][ni][r] * alpha;
        long idx = row * (long)ldc + col;
        if (mode == 0)
          Cf[idx] = a;
        else if (mode == 1)
          Cbf[idx] = f2bf(a);
        else if (mode == 2)
          Cf[idx] = a + residual[idx];
        else if (mode == 3)
          Cf[idx] += rowScale[row * scaleStride] * a;
        else
          Cf[idx] = residual[idx] + rowScale[row * scaleStride] * a;
      }
    }
  }
}

// ---------------------------------------------------------------------------
// RMSNorm (fp32 in) -> bf16 out.  One block (8 waves) per token, D = 1024.
// ---------------------------------------------------------------------------
__global__ __launch_bounds__(256) void rmsnorm_bf16(
    const float* __restrict__ x, const float* __restrict__ w,
    unsigned short* __restrict__ out, int D) {
  long t = blockIdx.x;
  const float* xr = x + t * (long)D;
  float ss = 0.f;
  for (int k = threadIdx.x; k < D; k += 256) {
    float v = xr[k];
    ss += v * v;
  }
#pragma unroll
  for (int o = 16; o > 0; o >>= 1) ss += __shfl_xor(ss, o, 32);
  __shared__ float red[8];
  int lane = threadIdx.x & 31, wv = threadIdx.x >> 5;
  if (lane == 0) red[wv] = ss;
  __syncthreads();
  float tot = 0.f;
#pragma unroll
  for (int i = 0; i < 8; ++i) tot += red[i];
  float sc = rsqrtf(tot / (float)D + 1e-6f);
  for (int k = threadIdx.x; k < D; k += 256)
    out[t * (long)D + k] = f2bf(xr[k] * sc * w[k]);
}

// ---------------------------------------------------------------------------
// Causal softmax over one pre-scaled score row; writes bf16 probabilities.
// ---------------------------------------------------------------------------
__global__ __launch_bounds__(256) void softmax_causal(
    const float* __restrict__ sc, unsigned short* __restrict__ P, int S) {
  long q = blockIdx.x;
  const float* row = sc + q * (long)S;
  unsigned short* pr = P + q * (long)S;
  int n = (int)q + 1;
  __shared__ float red[8];
  int lane = threadIdx.x & 31, wv = threadIdx.x >> 5;

  float mx = -3.4e38f;
  for (int k = threadIdx.x; k < n; k += 256) mx = fmaxf(mx, row[k]);
#pragma unroll
  for (int o = 16; o > 0; o >>= 1) mx = fmaxf(mx, __shfl_xor(mx, o, 32));
  if (lane == 0) red[wv] = mx;
  __syncthreads();
  float bmax = -3.4e38f;
#pragma unroll
  for (int i = 0; i < 8; ++i) bmax = fmaxf(bmax, red[i]);
  __syncthreads();

  float s = 0.f;
  for (int k = threadIdx.x; k < n; k += 256) s += __expf(row[k] - bmax);
#pragma unroll
  for (int o = 16; o > 0; o >>= 1) s += __shfl_xor(s, o, 32);
  if (lane == 0) red[wv] = s;
  __syncthreads();
  float tot = 0.f;
#pragma unroll
  for (int i = 0; i < 8; ++i) tot += red[i];
  float inv = 1.f / tot;

  for (int k = threadIdx.x; k < S; k += 256)
    pr[k] = (k < n) ? f2bf(__expf(row[k] - bmax) * inv) : (unsigned short)0;
}

// ---------------------------------------------------------------------------
// Router logits + softmax + top-2 gating, one wave per token (E = 8).
// ---------------------------------------------------------------------------
__global__ __launch_bounds__(256) void router_gate(
    const unsigned short* __restrict__ hn, const float* __restrict__ rw,
    float* __restrict__ combine, int T, int D) {
  int t = blockIdx.x * 8 + (threadIdx.x >> 5);
  int lane = threadIdx.x & 31;
  if (t >= T) return;
  float acc[8];
#pragma unroll
  for (int e = 0; e < 8; ++e) acc[e] = 0.f;
  const unsigned short* xr = hn + (long)t * D;
  for (int k = lane; k < D; k += 32) {
    float xv = bf2f(xr[k]);
    const float* r = rw + (long)k * 8;
#pragma unroll
    for (int e = 0; e < 8; ++e) acc[e] += xv * r[e];
  }
#pragma unroll
  for (int o = 16; o > 0; o >>= 1)
#pragma unroll
    for (int e = 0; e < 8; ++e) acc[e] += __shfl_xor(acc[e], o, 32);
  if (lane == 0) {
    float mx = acc[0];
#pragma unroll
    for (int e = 1; e < 8; ++e) mx = fmaxf(mx, acc[e]);
    float p[8], s = 0.f;
#pragma unroll
    for (int e = 0; e < 8; ++e) {
      p[e] = __expf(acc[e] - mx);
      s += p[e];
    }
#pragma unroll
    for (int e = 0; e < 8; ++e) p[e] /= s;
    int i1 = 0;
#pragma unroll
    for (int e = 1; e < 8; ++e)
      if (p[e] > p[i1]) i1 = e;
    int i2 = (i1 == 0) ? 1 : 0;
#pragma unroll
    for (int e = 0; e < 8; ++e)
      if (e != i1 && p[e] > p[i2]) i2 = e;
    float g = p[i1] + p[i2];
    float* c = combine + (long)t * 8;
#pragma unroll
    for (int e = 0; e < 8; ++e) c[e] = 0.f;
    c[i1] = p[i1] / g;
    c[i2] = p[i2] / g;
  }
}

// ---------------------------------------------------------------------------
// Elementwise helpers.
// ---------------------------------------------------------------------------
__global__ __launch_bounds__(256) void f32_to_bf16_kernel(
    const float* __restrict__ in, unsigned short* __restrict__ out, long n) {
  long i = (long)blockIdx.x * blockDim.x + threadIdx.x;
  if (i < n) out[i] = f2bf(in[i]);
}

__global__ __launch_bounds__(256) void silu_mul_bf16(
    const float* __restrict__ g1, const float* __restrict__ g3,
    unsigned short* __restrict__ out, long n) {
  long i = (long)blockIdx.x * blockDim.x + threadIdx.x;
  if (i < n) {
    float a = g1[i];
    float s = a / (1.f + __expf(-a));
    out[i] = f2bf(s * g3[i]);
  }
}

// ---------------------------------------------------------------------------
// Host orchestration.
// ---------------------------------------------------------------------------
extern "C" void kernel_launch(void* const* d_in, const int* in_sizes, int n_in,
                              void* d_out, int out_size, void* d_ws,
                              size_t ws_size, hipStream_t stream) {
  (void)in_sizes; (void)n_in; (void)out_size; (void)ws_size;
  const int Bb = 2, S = 2048, D = 1024, H = 16, HD = 64, E = 8, F = 1024;
  const int T = Bb * S;  // 4096 tokens

  const float* x        = (const float*)d_in[0];
  const float* attn_w   = (const float*)d_in[1];
  const float* wq       = (const float*)d_in[2];
  const float* wk       = (const float*)d_in[3];
  const float* wv       = (const float*)d_in[4];
  const float* wo       = (const float*)d_in[5];
  const float* ffn_w    = (const float*)d_in[6];
  const float* router_w = (const float*)d_in[7];
  const float* w1       = (const float*)d_in[8];
  const float* w3       = (const float*)d_in[9];
  const float* w2       = (const float*)d_in[10];
  float* out = (float*)d_out;

  // ---- workspace carve-up (~120 MB) ----
  char* w = (char*)d_ws;
  const size_t TD = (size_t)T * D;
  const size_t SS = (size_t)S * S;
  unsigned short* A_bf = (unsigned short*)w; w += TD * 2;  // xn, then hn
  unsigned short* q_bf = (unsigned short*)w; w += TD * 2;  // also h1 in MoE
  unsigned short* k_bf = (unsigned short*)w; w += TD * 2;
  unsigned short* v_bf = (unsigned short*)w; w += TD * 2;
  float* scores = (float*)w;                 w += SS * 4;  // also o_bf later
  unsigned short* P_bf = (unsigned short*)w; w += SS * 2;
  float* attn_out = (float*)w;               w += TD * 4;
  float* hbuf = (float*)w;                   w += TD * 4;
  float* G1 = (float*)w;                     w += TD * 4;
  float* G3 = (float*)w;                     w += TD * 4;
  float* combine = (float*)w;                w += (size_t)T * E * 4;
  unsigned short* o_bf = (unsigned short*)scores;  // reuse after attention
  unsigned short* h1_bf = q_bf;                    // reuse in MoE

  dim3 blk(256);
  dim3 gProj(D / BN, T / BM);      // (8, 32)
  dim3 gScore(S / BN, S / BM);     // (16, 16)
  dim3 gPV(1, S / BM);             // (1, 16)
  long nTD = (long)TD;
  dim3 gEw((unsigned)((nTD + 255) / 256));

  // 1) xn = rmsnorm(x) -> bf16
  rmsnorm_bf16<<<T, blk, 0, stream>>>(x, attn_w, A_bf, D);

  // 2) q, k, v projections (fp32 weights converted while staging)
  gemm_wmma<1><<<gProj, blk, 0, stream>>>(A_bf, D, wq, D, T, D, D, 1.f,
                                          nullptr, q_bf, D, nullptr, nullptr, 0, 1);
  gemm_wmma<1><<<gProj, blk, 0, stream>>>(A_bf, D, wk, D, T, D, D, 1.f,
                                          nullptr, k_bf, D, nullptr, nullptr, 0, 1);
  gemm_wmma<1><<<gProj, blk, 0, stream>>>(A_bf, D, wv, D, T, D, D, 1.f,
                                          nullptr, v_bf, D, nullptr, nullptr, 0, 1);

  // 3) attention per (batch, head): S = QK^T/sqrt(HD); P = causal softmax; O = P V
  const float inv_sqrt_hd = 0.125f;
  for (int b = 0; b < Bb; ++b) {
    for (int h = 0; h < H; ++h) {
      size_t base = (size_t)b * S * D + (size_t)h * HD;
      gemm_wmma<2><<<gScore, blk, 0, stream>>>(
          q_bf + base, D, k_bf + base, D, S, S, HD, inv_sqrt_hd, scores,
          nullptr, S, nullptr, nullptr, 0, 0);
      softmax_causal<<<S, blk, 0, stream>>>(scores, P_bf, S);
      gemm_wmma<0><<<gPV, blk, 0, stream>>>(
          P_bf, S, v_bf + base, D, S, HD, S, 1.f, attn_out + base, nullptr, D,
          nullptr, nullptr, 0, 0);
    }
  }

  // 4) h = x + (o @ wo)
  f32_to_bf16_kernel<<<gEw, blk, 0, stream>>>(attn_out, o_bf, nTD);
  gemm_wmma<1><<<gProj, blk, 0, stream>>>(o_bf, D, wo, D, T, D, D, 1.f, hbuf,
                                          nullptr, D, x, nullptr, 0, 2);

  // 5) hn = rmsnorm(h); router top-2 gating
  rmsnorm_bf16<<<T, blk, 0, stream>>>(hbuf, ffn_w, A_bf, D);
  router_gate<<<T / 8, blk, 0, stream>>>(A_bf, router_w, combine, T, D);

  // 6) MoE experts (dense-equivalent dispatch, accumulate into out)
  for (int e = 0; e < E; ++e) {
    const float* w1e = w1 + (size_t)e * D * F;
    const float* w3e = w3 + (size_t)e * D * F;
    const float* w2e = w2 + (size_t)e * F * D;
    gemm_wmma<1><<<gProj, blk, 0, stream>>>(A_bf, D, w1e, F, T, F, D, 1.f, G1,
                                            nullptr, F, nullptr, nullptr, 0, 0);
    gemm_wmma<1><<<gProj, blk, 0, stream>>>(A_bf, D, w3e, F, T, F, D, 1.f, G3,
                                            nullptr, F, nullptr, nullptr, 0, 0);
    silu_mul_bf16<<<gEw, blk, 0, stream>>>(G1, G3, h1_bf, nTD);
    // e == 0: out = h + gate*down ; e > 0: out += gate*down
    gemm_wmma<1><<<gProj, blk, 0, stream>>>(
        h1_bf, F, w2e, D, T, D, F, 1.f, out, nullptr, D, hbuf, combine + e, E,
        (e == 0) ? 4 : 3);
  }
}
